// GeniepathEncoder_29248727286046
// MI455X (gfx1250) — matrix operand order repaired and checked by reading, not compiled, not run on hardware
//
#include <hip/hip_runtime.h>
#include <hip/hip_bf16.h>
#include <cstdint>
#include <cstddef>

// ---------------- problem constants ----------------
#define NHOPS 3
#define HDIM  256
#define FDIM  128
#define ODIM  128

static const int kNSRC[NHOPS] = {60000, 30000, 15000};
static const int kNDST[NHOPS] = {30000, 15000, 7500};
static const int kNE[NHOPS]   = {480000, 240000, 120000};

// ---------------- types ----------------
typedef __attribute__((ext_vector_type(16))) __bf16 v16bf;
typedef __attribute__((ext_vector_type(8)))  float  v8f;
typedef __attribute__((ext_vector_type(4)))  unsigned int u32x4;

union FragB16 { unsigned int u[8]; u32x4 q[2]; v16bf v; };

__device__ __forceinline__ unsigned short f32_to_bf16(float f) {
  unsigned int u = __float_as_uint(f);
  u += 0x7fffu + ((u >> 16) & 1u);   // round to nearest even
  return (unsigned short)(u >> 16);
}
__device__ __forceinline__ float bf16_to_f32(unsigned short h) {
  return __uint_as_float(((unsigned int)h) << 16);
}
// order-preserving float -> uint map for atomicMax-based segment max
__device__ __forceinline__ unsigned int ford(float f) {
  unsigned int u = __float_as_uint(f);
  return (u & 0x80000000u) ? ~u : (u | 0x80000000u);
}
__device__ __forceinline__ float funord(unsigned int u) {
  return __uint_as_float((u & 0x80000000u) ? (u & 0x7fffffffu) : ~u);
}
__device__ __forceinline__ float sigmoidf_(float x) { return 1.0f / (1.0f + expf(-x)); }

// ---------------- generic helpers ----------------
__global__ void gp_zero_f32(float* p, size_t n) {
  size_t i = (size_t)blockIdx.x * blockDim.x + threadIdx.x;
  size_t stride = (size_t)gridDim.x * blockDim.x;
  for (; i < n; i += stride) p[i] = 0.0f;
}
__global__ void gp_zero_u32(unsigned int* p, size_t n) {
  size_t i = (size_t)blockIdx.x * blockDim.x + threadIdx.x;
  size_t stride = (size_t)gridDim.x * blockDim.x;
  for (; i < n; i += stride) p[i] = 0u;   // == ford(very negative); every dst has a self loop
}
__global__ void gp_cvt_bf16(const float* __restrict__ s, unsigned short* __restrict__ d, size_t n) {
  size_t i = (size_t)blockIdx.x * blockDim.x + threadIdx.x;
  size_t stride = (size_t)gridDim.x * blockDim.x;
  for (; i < n; i += stride) d[i] = f32_to_bf16(s[i]);
}
// dst[C x R] = bf16(src[R x C]^T)
__global__ void gp_transpose_bf16(const float* __restrict__ s, unsigned short* __restrict__ d,
                                  int R, int C) {
  int i = blockIdx.x * blockDim.x + threadIdx.x;
  if (i >= R * C) return;
  int r = i / C, c = i % C;
  d[(size_t)c * R + r] = f32_to_bf16(s[i]);
}
// Wihh^T row-major [1024 x 512]: row n = [W_ih[n][0:256] | W_hh[n][0:256]]
__global__ void gp_build_wihh(const float* __restrict__ Wih, const float* __restrict__ Whh,
                              unsigned short* __restrict__ d) {
  int i = blockIdx.x * blockDim.x + threadIdx.x;
  if (i >= 4 * HDIM * 2 * HDIM) return;
  int n = i / (2 * HDIM), k = i % (2 * HDIM);
  float v = (k < HDIM) ? Wih[(size_t)n * HDIM + k] : Whh[(size_t)n * HDIM + (k - HDIM)];
  d[i] = f32_to_bf16(v);
}
// vd[hop][k] = sum_j Wdst[hop][k][j] * att_dst[hop][j]
__global__ void gp_vd(const float* __restrict__ Wdst, const float* __restrict__ att_dst,
                      float* __restrict__ vd) {
  int hop = blockIdx.x, k = threadIdx.x;
  const float* W = Wdst + (size_t)hop * HDIM * HDIM + (size_t)k * HDIM;
  const float* a = att_dst + (size_t)hop * HDIM;
  float s = 0.0f;
  for (int j = 0; j < HDIM; ++j) s += W[j] * a[j];
  vd[hop * HDIM + k] = s;
}
// out[r] = dot(bf16 row r of X, vec)  — one wave32 per row, coalesced
__global__ void gp_rowdot(const unsigned short* __restrict__ X, const float* __restrict__ vec,
                          float* __restrict__ out, int nrows) {
  int w = (blockIdx.x * blockDim.x + threadIdx.x) >> 5;
  int lane = threadIdx.x & 31;
  if (w >= nrows) return;
  const unsigned short* row = X + (size_t)w * HDIM;
  float s = 0.0f;
  #pragma unroll
  for (int k = lane; k < HDIM; k += 32) s += bf16_to_f32(row[k]) * vec[k];
  #pragma unroll
  for (int off = 16; off > 0; off >>= 1) s += __shfl_down(s, off);
  if (lane == 0) out[w] = s;
}

// ---------------- bf16 WMMA GEMM: C[MxN] = A[MxK] @ Bt[NxK]^T (+bias) ----------------
// One wave computes a 16x64 strip; v_wmma_f32_16x16x32_bf16, f32 accumulate.
// Out-of-range M rows are handled by CLAMPING the A row index (A row m feeds only
// D row m, and D rows >= M are never stored), so the inner loop is branch-free and
// all fragment loads are b128.
template <bool OUT_BF16, bool HAS_BIAS>
__global__ __launch_bounds__(256)
void gp_gemm_bf16(const unsigned short* __restrict__ A,
                  const unsigned short* __restrict__ Bt,
                  float* __restrict__ Cf, unsigned short* __restrict__ Cb,
                  const float* __restrict__ bias, int M, int N, int K) {
  int wave = (blockIdx.x * blockDim.x + threadIdx.x) >> 5;
  int lane = threadIdx.x & 31;
  int nt64 = N >> 6;
  int tilesM = (M + 15) >> 4;
  if (wave >= tilesM * nt64) return;            // wave-uniform: EXEC stays all-ones
  int tn = wave % nt64;
  int tm = wave / nt64;
  int half = lane >> 4;
  int l15 = lane & 15;
  int m = tm * 16 + l15;
  if (m >= M) m = M - 1;                         // clamp: garbage rows never stored
  int nbase = tn * 64 + l15;
  int kbaseA = half * 16;                        // 16-bit A layout (sequential halves)
  int kbaseB = half * 16;                        // 16-bit B layout (sequential halves, same as A)

  v8f acc0 = {}, acc1 = {}, acc2 = {}, acc3 = {};
  const unsigned short* arow = A + (size_t)m * K + kbaseA;
  const unsigned short* brow0 = Bt + (size_t)(nbase +  0) * K + kbaseB;
  const unsigned short* brow1 = Bt + (size_t)(nbase + 16) * K + kbaseB;
  const unsigned short* brow2 = Bt + (size_t)(nbase + 32) * K + kbaseB;
  const unsigned short* brow3 = Bt + (size_t)(nbase + 48) * K + kbaseB;

  for (int k0 = 0; k0 < K; k0 += 32) {
    FragB16 a, b0, b1, b2, b3;
    a.q[0]  = *(const u32x4*)(arow + k0);        // K = kbaseA .. +7
    a.q[1]  = *(const u32x4*)(arow + k0 + 8);    // K = kbaseA+8 .. +15
    b0.q[0] = *(const u32x4*)(brow0 + k0);
    b0.q[1] = *(const u32x4*)(brow0 + k0 + 8);
    b1.q[0] = *(const u32x4*)(brow1 + k0);
    b1.q[1] = *(const u32x4*)(brow1 + k0 + 8);
    b2.q[0] = *(const u32x4*)(brow2 + k0);
    b2.q[1] = *(const u32x4*)(brow2 + k0 + 8);
    b3.q[0] = *(const u32x4*)(brow3 + k0);
    b3.q[1] = *(const u32x4*)(brow3 + k0 + 8);
    acc0 = __builtin_amdgcn_wmma_f32_16x16x32_bf16(false, a.v, false, b0.v, (short)0, acc0, false, false);
    acc1 = __builtin_amdgcn_wmma_f32_16x16x32_bf16(false, a.v, false, b1.v, (short)0, acc1, false, false);
    acc2 = __builtin_amdgcn_wmma_f32_16x16x32_bf16(false, a.v, false, b2.v, (short)0, acc2, false, false);
    acc3 = __builtin_amdgcn_wmma_f32_16x16x32_bf16(false, a.v, false, b3.v, (short)0, acc3, false, false);
  }

  #pragma unroll
  for (int nt = 0; nt < 4; ++nt) {
    v8f acc = (nt == 0) ? acc0 : (nt == 1) ? acc1 : (nt == 2) ? acc2 : acc3;
    int n = nbase + nt * 16;
    float bv = HAS_BIAS ? bias[n] : 0.0f;
    #pragma unroll
    for (int r = 0; r < 8; ++r) {
      int mm = tm * 16 + r + half * 8;           // ISA f32 C/D layout
      if (mm < M) {
        float v = acc[r] + bv;
        if (OUT_BF16) Cb[(size_t)mm * N + n] = f32_to_bf16(v);
        else          Cf[(size_t)mm * N + n] = v;
      }
    }
  }
}

// ---------------- GAT edge phases ----------------
__global__ void gp_edge_p1(const int* __restrict__ src, const int* __restrict__ dst,
                           int E, int n_dst, const float* __restrict__ aS,
                           const float* __restrict__ aD, float* __restrict__ ealpha,
                           unsigned int* __restrict__ amaxU) {
  int i = blockIdx.x * blockDim.x + threadIdx.x;
  int tot = E + n_dst;
  if (i >= tot) return;
  int s = (i < E) ? src[i] : (i - E);
  int d = (i < E) ? dst[i] : (i - E);
  float a = aS[s] + aD[d];
  a = (a > 0.0f) ? a : 0.2f * a;                 // leaky relu
  ealpha[i] = a;
  atomicMax(&amaxU[d], ford(a));
}
__global__ void gp_edge_p2(const int* __restrict__ dst, int E, int n_dst,
                           const unsigned int* __restrict__ amaxU,
                           float* __restrict__ ealpha, float* __restrict__ denom) {
  int i = blockIdx.x * blockDim.x + threadIdx.x;
  int tot = E + n_dst;
  if (i >= tot) return;
  int d = (i < E) ? dst[i] : (i - E);
  float e = expf(ealpha[i] - funord(amaxU[d]));
  ealpha[i] = e;
  atomicAdd(&denom[d], e);
}
// one wave32 per edge: agg[d][:] += w * xs[s][:]
__global__ void gp_edge_p3(const int* __restrict__ src, const int* __restrict__ dst,
                           int E, int n_dst, const float* __restrict__ ealpha,
                           const float* __restrict__ denom,
                           const unsigned short* __restrict__ xsB, float* __restrict__ agg) {
  int w = (blockIdx.x * blockDim.x + threadIdx.x) >> 5;
  int lane = threadIdx.x & 31;
  int tot = E + n_dst;
  if (w >= tot) return;
  int s = (w < E) ? src[w] : (w - E);
  int d = (w < E) ? dst[w] : (w - E);
  float wt = ealpha[w] / denom[d];
  const unsigned short* xrow = xsB + (size_t)s * HDIM;
  float* arow = agg + (size_t)d * HDIM;
  #pragma unroll
  for (int k = lane; k < HDIM; k += 32)
    atomicAdd(arow + k, wt * bf16_to_f32(xrow[k]));
}

// xb = tanh(agg + gat_b); xcat row = [bf16(xb) | bf16(h_prev)]
__global__ void gp_xbcat(const float* __restrict__ agg, const float* __restrict__ gb,
                         const float* __restrict__ hF, unsigned short* __restrict__ xcat,
                         int n_dst) {
  int i = blockIdx.x * blockDim.x + threadIdx.x;
  if (i >= n_dst * HDIM) return;
  int r = i >> 8, k = i & (HDIM - 1);
  float xb = tanhf(agg[i] + gb[k]);
  xcat[(size_t)r * (2 * HDIM) + k] = f32_to_bf16(xb);
  xcat[(size_t)r * (2 * HDIM) + HDIM + k] = f32_to_bf16(hF[i]);
}
// LSTM elementwise; writes c, h (f32) and next-hop input x (bf16)
__global__ void gp_lstm(const float* __restrict__ gates, float* __restrict__ cF,
                        float* __restrict__ hF, unsigned short* __restrict__ xB, int n_dst) {
  int i = blockIdx.x * blockDim.x + threadIdx.x;
  if (i >= n_dst * HDIM) return;
  int r = i >> 8, k = i & (HDIM - 1);
  const float* g = gates + (size_t)r * 4 * HDIM;
  float ig = sigmoidf_(g[k]);
  float fg = sigmoidf_(g[HDIM + k]);
  float gg = tanhf(g[2 * HDIM + k]);
  float og = sigmoidf_(g[3 * HDIM + k]);
  float cn = fg * cF[i] + ig * gg;
  float hn = og * tanhf(cn);
  cF[i] = cn;
  hF[i] = hn;
  xB[i] = f32_to_bf16(hn);
}

// ---------------- host-side orchestration ----------------
static inline int gemm_blocks(int M, int N) {
  int waves = ((M + 15) / 16) * (N / 64);
  return (waves + 7) / 8;   // 8 waves / 256-thread block
}

extern "C" void kernel_launch(void* const* d_in, const int* in_sizes, int n_in,
                              void* d_out, int out_size, void* d_ws, size_t ws_size,
                              hipStream_t stream) {
  (void)in_sizes; (void)n_in; (void)out_size; (void)ws_size;
  const float* feats   = (const float*)d_in[0];
  const float* Win     = (const float*)d_in[1];
  const float* b_in    = (const float*)d_in[2];
  const float* Wsrc    = (const float*)d_in[3];
  const float* Wdst    = (const float*)d_in[4];
  const float* att_src = (const float*)d_in[5];
  const float* att_dst = (const float*)d_in[6];
  const float* gat_b   = (const float*)d_in[7];
  const float* W_ih    = (const float*)d_in[8];
  const float* W_hh    = (const float*)d_in[9];
  const float* Wout    = (const float*)d_in[10];
  const float* b_out   = (const float*)d_in[11];
  const int* srcs[3] = {(const int*)d_in[12], (const int*)d_in[14], (const int*)d_in[16]};
  const int* dsts[3] = {(const int*)d_in[13], (const int*)d_in[15], (const int*)d_in[17]};

  // workspace carve-up (~335 MB)
  char* cur = (char*)d_ws;
  auto carve = [&](size_t bytes) -> void* {
    void* r = (void*)cur;
    cur += (bytes + 255) & ~(size_t)255;
    return r;
  };
  unsigned short* featsB = (unsigned short*)carve((size_t)60000 * FDIM * 2);
  unsigned short* xB     = (unsigned short*)carve((size_t)60000 * HDIM * 2);
  unsigned short* xsB    = (unsigned short*)carve((size_t)60000 * HDIM * 2);
  unsigned short* xcat   = (unsigned short*)carve((size_t)30000 * 2 * HDIM * 2);
  float* gates  = (float*)carve((size_t)30000 * 4 * HDIM * 4);
  float* agg    = (float*)carve((size_t)30000 * HDIM * 4);
  float* hF     = (float*)carve((size_t)30000 * HDIM * 4);
  float* cF     = (float*)carve((size_t)30000 * HDIM * 4);
  float* aS     = (float*)carve((size_t)60000 * 4);
  float* aD     = (float*)carve((size_t)30000 * 4);
  unsigned int* amaxU = (unsigned int*)carve((size_t)30000 * 4);
  float* denom  = (float*)carve((size_t)30000 * 4);
  float* ealpha = (float*)carve((size_t)510000 * 4);
  unsigned short* WinT  = (unsigned short*)carve((size_t)HDIM * FDIM * 2);
  unsigned short* WsrcT = (unsigned short*)carve((size_t)NHOPS * HDIM * HDIM * 2);
  unsigned short* WihhT = (unsigned short*)carve((size_t)NHOPS * 4 * HDIM * 2 * HDIM * 2);
  unsigned short* WoutT = (unsigned short*)carve((size_t)ODIM * HDIM * 2);
  float* vdBuf  = (float*)carve((size_t)NHOPS * HDIM * 4);

  const int T = 256;
  // ---- weight prep (bf16, pre-transposed N x K) ----
  gp_cvt_bf16<<<(60000 * FDIM + T - 1) / T, T, 0, stream>>>(feats, featsB, (size_t)60000 * FDIM);
  gp_transpose_bf16<<<(FDIM * HDIM + T - 1) / T, T, 0, stream>>>(Win, WinT, FDIM, HDIM);
  for (int i = 0; i < NHOPS; ++i)
    gp_transpose_bf16<<<(HDIM * HDIM + T - 1) / T, T, 0, stream>>>(
        Wsrc + (size_t)i * HDIM * HDIM, WsrcT + (size_t)i * HDIM * HDIM, HDIM, HDIM);
  gp_transpose_bf16<<<(HDIM * ODIM + T - 1) / T, T, 0, stream>>>(Wout, WoutT, HDIM, ODIM);
  for (int i = 0; i < NHOPS; ++i)
    gp_build_wihh<<<(4 * HDIM * 2 * HDIM + T - 1) / T, T, 0, stream>>>(
        W_ih + (size_t)i * 4 * HDIM * HDIM, W_hh + (size_t)i * 4 * HDIM * HDIM,
        WihhT + (size_t)i * 4 * HDIM * 2 * HDIM);
  gp_vd<<<NHOPS, HDIM, 0, stream>>>(Wdst, att_dst, vdBuf);
  gp_zero_f32<<<((size_t)30000 * HDIM + T - 1) / T, T, 0, stream>>>(hF, (size_t)30000 * HDIM);
  gp_zero_f32<<<((size_t)30000 * HDIM + T - 1) / T, T, 0, stream>>>(cF, (size_t)30000 * HDIM);

  // ---- input projection: x = feats @ Win + b_in  -> bf16 xB ----
  gp_gemm_bf16<true, true><<<gemm_blocks(60000, HDIM), T, 0, stream>>>(
      featsB, WinT, nullptr, xB, b_in, 60000, HDIM, FDIM);

  // ---- hops ----
  for (int i = 0; i < NHOPS; ++i) {
    int n_src = kNSRC[i], n_dst = kNDST[i], E = kNE[i];
    int tot = E + n_dst;

    // xs = x @ Wsrc[i]  (bf16 out)
    gp_gemm_bf16<true, false><<<gemm_blocks(n_src, HDIM), T, 0, stream>>>(
        xB, WsrcT + (size_t)i * HDIM * HDIM, nullptr, xsB, nullptr, n_src, HDIM, HDIM);
    // alpha_s = xs . att_src[i] ; alpha_d = x[:n_dst] . (Wdst[i] @ att_dst[i])
    gp_rowdot<<<(n_src * 32 + T - 1) / T, T, 0, stream>>>(xsB, att_src + (size_t)i * HDIM, aS, n_src);
    gp_rowdot<<<(n_dst * 32 + T - 1) / T, T, 0, stream>>>(xB, vdBuf + (size_t)i * HDIM, aD, n_dst);

    gp_zero_u32<<<(n_dst + T - 1) / T, T, 0, stream>>>(amaxU, (size_t)n_dst);
    gp_zero_f32<<<(n_dst + T - 1) / T, T, 0, stream>>>(denom, (size_t)n_dst);
    gp_zero_f32<<<((size_t)n_dst * HDIM + T - 1) / T, T, 0, stream>>>(agg, (size_t)n_dst * HDIM);

    gp_edge_p1<<<(tot + T - 1) / T, T, 0, stream>>>(srcs[i], dsts[i], E, n_dst, aS, aD, ealpha, amaxU);
    gp_edge_p2<<<(tot + T - 1) / T, T, 0, stream>>>(dsts[i], E, n_dst, amaxU, ealpha, denom);
    gp_edge_p3<<<(tot + 7) / 8, T, 0, stream>>>(srcs[i], dsts[i], E, n_dst, ealpha, denom, xsB, agg);

    // xb = tanh(agg + b); xcat = [xb | h]
    gp_xbcat<<<((size_t)n_dst * HDIM + T - 1) / T, T, 0, stream>>>(
        agg, gat_b + (size_t)i * HDIM, hF, xcat, n_dst);
    // gates = [xb|h] @ [W_ih|W_hh]^T  (fused, K=512, N=1024, f32 out)
    gp_gemm_bf16<false, false><<<gemm_blocks(n_dst, 4 * HDIM), T, 0, stream>>>(
        xcat, WihhT + (size_t)i * 4 * HDIM * 2 * HDIM, gates, nullptr, nullptr,
        n_dst, 4 * HDIM, 2 * HDIM);
    // LSTM elementwise; writes next-hop x (bf16) into xB
    gp_lstm<<<((size_t)n_dst * HDIM + T - 1) / T, T, 0, stream>>>(gates, cF, hF, xB, n_dst);
  }

  // ---- output projection: out = h @ Wout + b_out (f32) ----
  gp_gemm_bf16<false, true><<<gemm_blocks(kNDST[NHOPS - 1], ODIM), T, 0, stream>>>(
      xB, WoutT, (float*)d_out, nullptr, b_out, kNDST[NHOPS - 1], ODIM, HDIM);
}
